// LMHSA_20804821582378
// MI455X (gfx1250) — matrix operand order repaired and checked by reading, not compiled, not run on hardware
//
#include <hip/hip_runtime.h>

typedef __attribute__((ext_vector_type(16))) _Float16 v16h;
typedef __attribute__((ext_vector_type(8)))  float    v8f;

#define WMMA_F16(a, b, c) \
  __builtin_amdgcn_wmma_f32_16x16x32_f16(false, (a), false, (b), (short)0, (c), false, false)

namespace {
constexpr int  kB = 4, kC = 256, kH = 96, kW = 96;
constexpr int  kDK = 32, kHD = 256;
constexpr int  kHW = kH * kW;                 // 9216
constexpr int  kNTOK = kB * kHW;              // 36864
constexpr int  kNW = kW / 4;                  // 24 vertical windows
constexpr int  kSV = kH * 4;                  // 384 tokens per window
constexpr int  kNPOOL = (kH / 8) * (kW / 8);  // 144 pooled tokens
constexpr int  kC1 = 192, kC2 = 64;
constexpr long kCHW = (long)kC * kH * kW;     // 2359296
constexpr long kNTOT = (long)kB * kCHW;       // 9437184
constexpr int  kGrpCnt = kB * kC * kH * 4;    // 393216 elems per vertical LN group
constexpr float kEPS = 1e-5f;
constexpr float kSCALE = 0.17677669529663687f; // 1/sqrt(32)
}

// max across each 16-lane DPP row (lanes 0-15 / 16-31 reduced independently);
// all lanes end up with their row's max. Pure VALU: v_mov_dpp + v_max, no LDS.
__device__ __forceinline__ float dpp_row_max16(float x) {
  float y;
  y = __int_as_float(__builtin_amdgcn_update_dpp(
        __float_as_int(x), __float_as_int(x), 0xB1 /*quad_perm(1,0,3,2)*/, 0xF, 0xF, true));
  x = fmaxf(x, y);
  y = __int_as_float(__builtin_amdgcn_update_dpp(
        __float_as_int(x), __float_as_int(x), 0x4E /*quad_perm(2,3,0,1)*/, 0xF, 0xF, true));
  x = fmaxf(x, y);
  y = __int_as_float(__builtin_amdgcn_update_dpp(
        __float_as_int(x), __float_as_int(x), 0x141 /*row_half_mirror*/, 0xF, 0xF, true));
  x = fmaxf(x, y);
  y = __int_as_float(__builtin_amdgcn_update_dpp(
        __float_as_int(x), __float_as_int(x), 0x140 /*row_mirror*/, 0xF, 0xF, true));
  x = fmaxf(x, y);
  return x;
}

// ---------------- f32 -> f16 weight conversion ----------------
__global__ void k_convert(const float* __restrict__ in, _Float16* __restrict__ out, int n) {
  int i = blockIdx.x * blockDim.x + threadIdx.x;
  if (i < n) out[i] = (_Float16)in[i];
}

// ---------------- global + per-window-group statistics of x ----------------
// grid.x = 24 groups * 32 sub-blocks
__global__ void k_stats_x(const float* __restrict__ x, float* __restrict__ stats) {
  __shared__ float sh1[256], sh2[256];
  const int n = blockIdx.x >> 5;
  const int sub = blockIdx.x & 31;
  float s = 0.f, s2 = 0.f;
  for (int q = sub * 256 + threadIdx.x; q < kGrpCnt; q += 32 * 256) {
    int dw = q & 3;
    int h = (q >> 2) % kH;
    int c = (q / (4 * kH)) % kC;
    int b = q / (4 * kH * kC);
    float v = x[(((long)(b * kC + c) * kH + h) * kW) + n * 4 + dw];
    s += v; s2 += v * v;
  }
  sh1[threadIdx.x] = s; sh2[threadIdx.x] = s2;
  __syncthreads();
  for (int o = 128; o > 0; o >>= 1) {
    if (threadIdx.x < o) {
      sh1[threadIdx.x] += sh1[threadIdx.x + o];
      sh2[threadIdx.x] += sh2[threadIdx.x + o];
    }
    __syncthreads();
  }
  if (threadIdx.x == 0) {
    atomicAdd(&stats[2 + 2 * n], sh1[0]);
    atomicAdd(&stats[3 + 2 * n], sh2[0]);
    atomicAdd(&stats[0], sh1[0]);
    atomicAdd(&stats[1], sh2[0]);
  }
}

// ---------------- build f16 token matrices in the three layouts ----------------
__global__ void k_build_tokens(const float* __restrict__ x, const float* __restrict__ stats,
                               _Float16* __restrict__ tokn_h, _Float16* __restrict__ tok_v,
                               _Float16* __restrict__ tokn_v) {
  const float mg = stats[0] / (float)kNTOT;
  const float vg = stats[1] / (float)kNTOT - mg * mg;
  const float rg = rsqrtf(vg + kEPS);
  for (long idx = (long)blockIdx.x * blockDim.x + threadIdx.x; idx < kNTOT;
       idx += (long)gridDim.x * blockDim.x) {
    int w = (int)(idx % kW);
    int h = (int)((idx / kW) % kH);
    int c = (int)((idx / (long)kHW) % kC);
    int b = (int)(idx / kCHW);
    float v = x[idx];
    int n = w >> 2;
    float mn = stats[2 + 2 * n] / (float)kGrpCnt;
    float vn = stats[3 + 2 * n] / (float)kGrpCnt - mn * mn;
    float rn = rsqrtf(vn + kEPS);
    tokn_h[(long)(b * kHW + h * kW + w) * kHD + c] = (_Float16)((v - mg) * rg);
    long rowv = (long)((b * kNW + n) * kSV + (h * 4 + (w & 3)));
    tok_v[rowv * kHD + c]  = (_Float16)v;
    tokn_v[rowv * kHD + c] = (_Float16)((v - mn) * rn);
  }
}

// ---------------- 8x8 average pool -> pooled tokens [B*144, 256] f16 ----------------
__global__ void k_pool(const float* __restrict__ x, _Float16* __restrict__ pooled) {
  int t = blockIdx.x * blockDim.x + threadIdx.x;
  if (t >= kB * kNPOOL * kC) return;
  int c = t % kC;
  int s = (t / kC) % kNPOOL;
  int b = t / (kC * kNPOOL);
  int i = s / 12, j = s % 12;
  const float* xp = x + ((long)(b * kC + c) * kH + i * 8) * kW + j * 8;
  float acc = 0.f;
  #pragma unroll
  for (int dh = 0; dh < 8; ++dh)
    #pragma unroll
    for (int dw = 0; dw < 8; ++dw)
      acc += xp[dh * kW + dw];
  pooled[(long)(b * kNPOOL + s) * kC + c] = (_Float16)(acc * (1.f / 64.f));
}

// ---------------- WMMA GEMM: C[M,N] = A[M,K] * B[K,N] + bias ----------------
// block = 256 threads = 8 wave32 waves; block tile 128x64; wave tile 32x32 (2x2 frags)
// As row-major [128][32] (A-frag reads contiguous); Bs transposed [64][32]
// (B-frag reads contiguous) -> all LDS fragment reads are 2x ds_load_b128.
template <bool OUT16, bool BOUND>
__global__ void __launch_bounds__(256) k_gemm_bias(
    const _Float16* __restrict__ A, const _Float16* __restrict__ B,
    const float* __restrict__ bias, void* __restrict__ Cptr,
    int M, int N, int K) {
  __shared__ _Float16 As[128 * 32];
  __shared__ _Float16 Bs[64 * 32];   // transposed: [col][k]
  const int tid = threadIdx.x;
  const int lane = tid & 31, wid = tid >> 5;
  const int wm = wid >> 1, wn = wid & 1;
  const int mlane = lane & 15, nlane = lane & 15, khi = lane >> 4;
  const int mBlk = blockIdx.x * 128;
  const int nBlk = blockIdx.y * 64;
  const int gcol0 = nBlk + wn * 32 + nlane;
  const float bias0 = bias[gcol0];
  const float bias1 = bias[gcol0 + 16];
  v8f acc[2][2] = {};
  const int nkc = K >> 5;
  for (int kc = 0; kc < nkc; ++kc) {
    __syncthreads();
    for (int i = tid; i < 128 * 32; i += 256) {
      int r = i >> 5, cc = i & 31;
      int gm = mBlk + r;
      As[i] = (!BOUND || gm < M) ? A[(long)gm * K + kc * 32 + cc] : (_Float16)0.f;
    }
    for (int i = tid; i < 32 * 64; i += 256) {
      int r = i >> 6, cc = i & 63;   // r = k row, cc = out col (coalesced global read)
      Bs[cc * 32 + r] = B[(long)(kc * 32 + r) * N + nBlk + cc];
    }
    __syncthreads();
    v16h a0, a1, b0, b1;
    #pragma unroll
    for (int e = 0; e < 16; ++e) {
      int kd = (khi << 4) + e;                           // A-frag K mapping (contiguous)
      a0[e] = As[(wm * 32 + mlane) * 32 + kd];
      a1[e] = As[(wm * 32 + 16 + mlane) * 32 + kd];
    }
    #pragma unroll
    for (int e = 0; e < 16; ++e) {
      int kd = (khi << 4) + e;                           // B-frag K mapping (contiguous)
      b0[e] = Bs[(wn * 32 + nlane) * 32 + kd];
      b1[e] = Bs[(wn * 32 + 16 + nlane) * 32 + kd];
    }
    acc[0][0] = WMMA_F16(a0, b0, acc[0][0]);
    acc[0][1] = WMMA_F16(a0, b1, acc[0][1]);
    acc[1][0] = WMMA_F16(a1, b0, acc[1][0]);
    acc[1][1] = WMMA_F16(a1, b1, acc[1][1]);
  }
  #pragma unroll
  for (int fi = 0; fi < 2; ++fi)
    #pragma unroll
    for (int i = 0; i < 8; ++i) {
      int gr = mBlk + wm * 32 + fi * 16 + i + 8 * khi;   // C-frag row map
      if (!BOUND || gr < M) {
        float v0 = acc[fi][0][i] + bias0;
        float v1 = acc[fi][1][i] + bias1;
        if (OUT16) {
          ((_Float16*)Cptr)[(long)gr * N + gcol0]      = (_Float16)v0;
          ((_Float16*)Cptr)[(long)gr * N + gcol0 + 16] = (_Float16)v1;
        } else {
          ((float*)Cptr)[(long)gr * N + gcol0]      = v0;
          ((float*)Cptr)[(long)gr * N + gcol0 + 16] = v1;
        }
      }
    }
}

// ---------------- flash attention chunk (32 keys, or 16-key tail when !FULL) ----------
template <bool FULL>
__device__ __forceinline__ void attn_chunk(
    int kb, int SkvPad, const _Float16* __restrict__ Ks, const _Float16* __restrict__ Vt,
    _Float16* __restrict__ myP, const v16h& qa, const v16h& ones, v8f& o0, v8f& o1,
    float (&rmax)[8], float (&rsum)[8], int mlane, int nlane, int khi) {
  v16h bk;
  #pragma unroll
  for (int e = 0; e < 16; ++e) {
    int kd = (khi << 4) + e;
    bk[e] = Ks[(kb + nlane) * 32 + kd];          // B[kdim][key] = K[key][kdim]
  }
  v8f s0 = {}; s0 = WMMA_F16(qa, bk, s0);
  v8f s1 = {};
  if (FULL) {
    v16h bk1;
    #pragma unroll
    for (int e = 0; e < 16; ++e) {
      int kd = (khi << 4) + e;
      bk1[e] = Ks[(kb + 16 + nlane) * 32 + kd];
    }
    s1 = WMMA_F16(qa, bk1, s1);
  }

  // online softmax: each v8f element i is row (i + 8*khi); columns live across 16 lanes.
  // Row max via DPP network (VALU only); row sum via WMMA with all-ones B below.
  float cr8[8];
  #pragma unroll
  for (int i = 0; i < 8; ++i) {
    float v0 = s0[i] * kSCALE;
    float v1 = FULL ? s1[i] * kSCALE : -1e30f;
    float t = dpp_row_max16(fmaxf(v0, v1));
    float mn = fmaxf(rmax[i], t);
    float cr = __expf(rmax[i] - mn);
    rmax[i] = mn;
    cr8[i] = cr;
    o0[i] *= cr; o1[i] *= cr;
    float p0 = __expf(v0 - mn);
    float p1 = FULL ? __expf(v1 - mn) : 0.f;
    int row = i + 8 * khi;
    myP[row * 32 + nlane] = (_Float16)p0;        // C-layout -> A-layout via LDS restage
    myP[row * 32 + 16 + nlane] = (_Float16)p1;
  }

  v16h pa;
  #pragma unroll
  for (int e = 0; e < 16; ++e) {
    int kd = (khi << 4) + e;
    pa[e] = myP[mlane * 32 + kd];                // same-wave LDS, in-order
  }
  // row sums of this P chunk, replicated across lanes, via P x ones
  v8f srow = {};
  srow = WMMA_F16(pa, ones, srow);
  #pragma unroll
  for (int i = 0; i < 8; ++i) rsum[i] = rsum[i] * cr8[i] + srow[i];

  v16h bv0, bv1;
  #pragma unroll
  for (int e = 0; e < 16; ++e) {
    int key = kb + (khi << 4) + e;               // contiguous; pad keys read zeros
    bv0[e] = Vt[nlane * SkvPad + key];
    bv1[e] = Vt[(16 + nlane) * SkvPad + key];
  }
  o0 = WMMA_F16(pa, bv0, o0);
  o1 = WMMA_F16(pa, bv1, o1);
}

// ---------------- flash-style WMMA attention ----------------
// grid: x = q-row-tiles-of-128 per group, y = head (8), z = group
// Q/K/V/O rows are [*, 256] f16, head slice = head*32.
// Ks: [Skv][32] (score B-frag contiguous over kd); Vt: [32][SkvPad] transposed +
// zero-padded keys (PV B-frag contiguous over key, no per-element clamp).
__global__ void __launch_bounds__(256) k_attention(
    const _Float16* __restrict__ Q, const _Float16* __restrict__ K,
    const _Float16* __restrict__ V, _Float16* __restrict__ O,
    int Skv, long qGroupStride, long kvGroupStride) {
  extern __shared__ _Float16 smem[];
  const int SkvPad = (Skv + 31) & ~31;
  _Float16* Ks  = smem;                          // [Skv][32]
  _Float16* Vt  = smem + (size_t)Skv * 32;       // [32][SkvPad]
  _Float16* Pst = Vt + (size_t)32 * SkvPad;      // 8 waves * 16x32 staging
  const int tid = threadIdx.x;
  const int lane = tid & 31, wid = tid >> 5;
  const int mlane = lane & 15, nlane = lane & 15, khi = lane >> 4;
  const int head = blockIdx.y;
  const long grp = blockIdx.z;
  const long qBase = grp * qGroupStride + (long)(blockIdx.x * 128 + wid * 16) * kHD + head * kDK;
  const long kvBase = grp * kvGroupStride + head * kDK;

  for (int i = tid; i < Skv * 32; i += 256) {
    int r = i >> 5, cc = i & 31;
    Ks[i] = K[kvBase + (long)r * kHD + cc];
    Vt[cc * SkvPad + r] = V[kvBase + (long)r * kHD + cc];
  }
  const int pad = SkvPad - Skv;
  for (int i = tid; i < pad * 32; i += 256) {
    int r = Skv + (i % pad);
    int cc = i / pad;
    Vt[cc * SkvPad + r] = (_Float16)0.f;
  }
  __syncthreads();

  v16h qa;
  #pragma unroll
  for (int e = 0; e < 16; ++e) {
    int kd = (khi << 4) + e;
    qa[e] = Q[qBase + (long)mlane * kHD + kd];
  }
  v16h ones;
  #pragma unroll
  for (int e = 0; e < 16; ++e) ones[e] = (_Float16)1.0f;

  float rmax[8], rsum[8];
  #pragma unroll
  for (int i = 0; i < 8; ++i) { rmax[i] = -1e30f; rsum[i] = 0.f; }
  v8f o0 = {}, o1 = {};
  _Float16* myP = Pst + wid * (16 * 32);

  const int nFull = Skv >> 5;                    // branch-free main loop
  for (int kc = 0; kc < nFull; ++kc)
    attn_chunk<true>(kc * 32, SkvPad, Ks, Vt, myP, qa, ones, o0, o1, rmax, rsum,
                     mlane, nlane, khi);
  if (Skv & 16)                                  // single 16-key tail (Skv=144 case)
    attn_chunk<false>(nFull * 32, SkvPad, Ks, Vt, myP, qa, ones, o0, o1, rmax, rsum,
                      mlane, nlane, khi);

  #pragma unroll
  for (int i = 0; i < 8; ++i) {
    float inv = __builtin_amdgcn_rcpf(rsum[i]);  // softmax denom: fast rcp is plenty
    int row = i + 8 * khi;
    O[qBase + (long)row * kHD + nlane]      = (_Float16)(o0[i] * inv);
    O[qBase + (long)row * kHD + 16 + nlane] = (_Float16)(o1[i] * inv);
  }
}

// ---------------- concat + residual + per-sample stats ----------------
__global__ void k_combine(const float* __restrict__ x, const float* __restrict__ ha,
                          const float* __restrict__ va, float* __restrict__ outp,
                          float* __restrict__ stats) {
  __shared__ float sh1[256], sh2[256];
  constexpr int nb = 2304;                        // 2359296 / 1024
  const int b = blockIdx.x / nb;
  const int sub = blockIdx.x % nb;
  float s = 0.f, s2 = 0.f;
  #pragma unroll
  for (int r = 0; r < 4; ++r) {
    int e = sub * 1024 + r * 256 + threadIdx.x;   // index within sample
    int c = e / kHW;
    int rem = e % kHW;
    int h = rem / kW;
    int w = rem % kW;
    float v;
    if (c < kC1) {
      v = ha[(long)(b * kHW + rem) * kC1 + c];
    } else {
      long rowv = (long)((b * kNW + (w >> 2)) * kSV + h * 4 + (w & 3));
      v = va[rowv * kC2 + (c - kC1)];
    }
    v += x[(long)b * kCHW + e];
    outp[(long)b * kCHW + e] = v;
    s += v; s2 += v * v;
  }
  sh1[threadIdx.x] = s; sh2[threadIdx.x] = s2;
  __syncthreads();
  for (int o = 128; o > 0; o >>= 1) {
    if (threadIdx.x < o) {
      sh1[threadIdx.x] += sh1[threadIdx.x + o];
      sh2[threadIdx.x] += sh2[threadIdx.x + o];
    }
    __syncthreads();
  }
  if (threadIdx.x == 0) {
    atomicAdd(&stats[64 + 2 * b], sh1[0]);
    atomicAdd(&stats[65 + 2 * b], sh2[0]);
  }
}

// ---------------- final per-sample layer norm ----------------
__global__ void k_final_ln(const float* __restrict__ outp, const float* __restrict__ stats,
                           float* __restrict__ out) {
  for (long idx = (long)blockIdx.x * blockDim.x + threadIdx.x; idx < kNTOT;
       idx += (long)gridDim.x * blockDim.x) {
    int b = (int)(idx / kCHW);
    float m = stats[64 + 2 * b] / (float)kCHW;
    float v = stats[65 + 2 * b] / (float)kCHW - m * m;
    out[idx] = (outp[idx] - m) * rsqrtf(v + kEPS);
  }
}

extern "C" void kernel_launch(void* const* d_in, const int* in_sizes, int n_in,
                              void* d_out, int out_size, void* d_ws, size_t ws_size,
                              hipStream_t stream) {
  (void)in_sizes; (void)n_in; (void)out_size; (void)ws_size;
  const float* x   = (const float*)d_in[0];
  const float* hWq = (const float*)d_in[1];
  const float* hbq = (const float*)d_in[2];
  const float* hWk = (const float*)d_in[3];
  const float* hbk = (const float*)d_in[4];
  const float* hWv = (const float*)d_in[5];
  const float* hbv = (const float*)d_in[6];
  const float* hWo = (const float*)d_in[7];
  const float* hbo = (const float*)d_in[8];
  const float* vWq = (const float*)d_in[9];
  const float* vbq = (const float*)d_in[10];
  const float* vWk = (const float*)d_in[11];
  const float* vbk = (const float*)d_in[12];
  const float* vWv = (const float*)d_in[13];
  const float* vbv = (const float*)d_in[14];
  const float* vWo = (const float*)d_in[15];
  const float* vbo = (const float*)d_in[16];

  char* base = (char*)d_ws;
  size_t off = 0;
  auto alloc = [&](size_t bytes) -> void* {
    void* p = base + off;
    off += (bytes + 255) & ~(size_t)255;
    return p;
  };
  float*     stats  = (float*)alloc(1024);
  _Float16*  tokn_h = (_Float16*)alloc((size_t)kNTOK * kHD * 2);
  _Float16*  tok_v  = (_Float16*)alloc((size_t)kNTOK * kHD * 2);
  _Float16*  tokn_v = (_Float16*)alloc((size_t)kNTOK * kHD * 2);
  _Float16*  pooled = (_Float16*)alloc((size_t)kB * kNPOOL * kC * 2);
  _Float16*  hQ     = (_Float16*)alloc((size_t)kNTOK * kHD * 2);
  _Float16*  hK     = (_Float16*)alloc((size_t)kB * kNPOOL * kHD * 2);
  _Float16*  hV     = (_Float16*)alloc((size_t)kB * kNPOOL * kHD * 2);
  _Float16*  vQ     = (_Float16*)alloc((size_t)kNTOK * kHD * 2);
  _Float16*  vK     = (_Float16*)alloc((size_t)kNTOK * kHD * 2);
  _Float16*  vV     = (_Float16*)alloc((size_t)kNTOK * kHD * 2);
  _Float16*  hO     = (_Float16*)alloc((size_t)kNTOK * kHD * 2);
  _Float16*  vO     = (_Float16*)alloc((size_t)kNTOK * kHD * 2);
  float*     ha     = (float*)alloc((size_t)kNTOK * kC1 * 4);
  float*     va     = (float*)alloc((size_t)kNTOK * kC2 * 4);
  float*     outp   = (float*)alloc((size_t)kNTOT * 4);
  _Float16*  w_hq   = (_Float16*)alloc(65536 * 2);
  _Float16*  w_hk   = (_Float16*)alloc(65536 * 2);
  _Float16*  w_hv   = (_Float16*)alloc(65536 * 2);
  _Float16*  w_ho   = (_Float16*)alloc(49152 * 2);
  _Float16*  w_vq   = (_Float16*)alloc(65536 * 2);
  _Float16*  w_vk   = (_Float16*)alloc(65536 * 2);
  _Float16*  w_vv   = (_Float16*)alloc(65536 * 2);
  _Float16*  w_vo   = (_Float16*)alloc(16384 * 2);

  hipMemsetAsync(stats, 0, 1024, stream);

  k_convert<<<256, 256, 0, stream>>>(hWq, w_hq, 65536);
  k_convert<<<256, 256, 0, stream>>>(hWk, w_hk, 65536);
  k_convert<<<256, 256, 0, stream>>>(hWv, w_hv, 65536);
  k_convert<<<192, 256, 0, stream>>>(hWo, w_ho, 49152);
  k_convert<<<256, 256, 0, stream>>>(vWq, w_vq, 65536);
  k_convert<<<256, 256, 0, stream>>>(vWk, w_vk, 65536);
  k_convert<<<256, 256, 0, stream>>>(vWv, w_vv, 65536);
  k_convert<<<64,  256, 0, stream>>>(vWo, w_vo, 16384);

  k_stats_x<<<dim3(24 * 32), 256, 0, stream>>>(x, stats);
  k_build_tokens<<<9216, 256, 0, stream>>>(x, stats, tokn_h, tok_v, tokn_v);
  k_pool<<<(kB * kNPOOL * kC + 255) / 256, 256, 0, stream>>>(x, pooled);

  // projections (A tokens f16, B weights f16, f32 accum, fused bias)
  k_gemm_bias<true,  false><<<dim3(288, 4), 256, 0, stream>>>(tokn_h, w_hq, hbq, hQ, kNTOK, 256, 256);
  k_gemm_bias<true,  true ><<<dim3(5, 4),   256, 0, stream>>>(pooled, w_hk, hbk, hK, kB * kNPOOL, 256, 256);
  k_gemm_bias<true,  true ><<<dim3(5, 4),   256, 0, stream>>>(pooled, w_hv, hbv, hV, kB * kNPOOL, 256, 256);
  k_gemm_bias<true,  false><<<dim3(288, 4), 256, 0, stream>>>(tokn_v, w_vq, vbq, vQ, kNTOK, 256, 256);
  k_gemm_bias<true,  false><<<dim3(288, 4), 256, 0, stream>>>(tok_v,  w_vk, vbk, vK, kNTOK, 256, 256);
  k_gemm_bias<true,  false><<<dim3(288, 4), 256, 0, stream>>>(tok_v,  w_vv, vbv, vV, kNTOK, 256, 256);

  // horizontal attention: groups = batch, Sq=9216, Skv=144 (SkvPad=160)
  size_t shH = ((size_t)144 * 32 + 32 * 160 + 8 * 16 * 32) * sizeof(_Float16);
  k_attention<<<dim3(72, 8, 4), 256, shH, stream>>>(hQ, hK, hV, hO, 144,
                                                    (long)kHW * kHD, (long)kNPOOL * kHD);
  // vertical attention: groups = b*24+n (96), Sq=Skv=384
  size_t shV = ((size_t)384 * 32 + 32 * 384 + 8 * 16 * 32) * sizeof(_Float16);
  k_attention<<<dim3(3, 8, 96), 256, shV, stream>>>(vQ, vK, vV, vO, 384,
                                                    (long)kSV * kHD, (long)kSV * kHD);

  // output projections (f32 out)
  k_gemm_bias<false, false><<<dim3(288, 3), 256, 0, stream>>>(hO, w_ho, hbo, ha, kNTOK, kC1, 256);
  k_gemm_bias<false, false><<<dim3(288, 1), 256, 0, stream>>>(vO, w_vo, vbo, va, kNTOK, kC2, 256);

  k_combine<<<kB * 2304, 256, 0, stream>>>(x, ha, va, outp, stats);
  k_final_ln<<<9216, 256, 0, stream>>>(outp, stats, (float*)d_out);
}
